// MultiHeadAttention_46076409151722
// MI455X (gfx1250) — matrix operand order repaired and checked by reading, not compiled, not run on hardware
//
#include <hip/hip_runtime.h>
#include <hip/hip_bf16.h>
#include <cstdint>

// Problem constants (from reference): B=2, T=2048, C=1024, H=16, HS=64
#define NH    16
#define HS    64
#define CDIM  1024
#define BATCH 2
#define TSEQ  2048
#define BT    (BATCH * TSEQ)

typedef __attribute__((ext_vector_type(16))) __bf16 v16bf;
typedef __attribute__((ext_vector_type(8)))  float  v8f;

union FragA { v16bf v; uint2 d2[4]; uint4 d4[2]; };
union FragB { v16bf v; uint4 d4[2]; };

static __device__ __forceinline__ unsigned short f2bf(float f) {
    union { float f; uint32_t u; } a; a.f = f;
    uint32_t r = a.u + 0x7FFFu + ((a.u >> 16) & 1u);   // round-to-nearest-even
    return (unsigned short)(r >> 16);
}

// A fragment: 16x32 bf16, row-major source (row stride ld), tile origin `tile`,
// K offset koff. Per CDNA5 layout: lane row = lane%16; VGPR0..3 hold
// K = half*8 + {0..7}; VGPR4..7 hold K = 16 + half*8 + {0..7}.
static __device__ __forceinline__ void load_a(FragA& f, const unsigned short* tile,
                                              int ld, int koff, int l16, int half) {
    const unsigned short* r = tile + (size_t)l16 * ld + koff + half * 8;
    f.d2[0] = *(const uint2*)(r);
    f.d2[1] = *(const uint2*)(r + 4);
    f.d2[2] = *(const uint2*)(r + 16);
    f.d2[3] = *(const uint2*)(r + 20);
}

// B fragment: 32x16 bf16 from a COLUMN-MAJOR source: element (k, n) lives at
// colmaj[n*ldn + k]. Lane column n = lane%16; elements 0..7 hold K = half*8 + 0..7,
// elements 8..15 hold K = 16 + half*8 + 0..7 -> two b128 loads.
static __device__ __forceinline__ void load_b(FragB& f, const unsigned short* colmaj,
                                              int ldn, int koff, int l16, int half) {
    const unsigned short* c = colmaj + (size_t)l16 * ldn + koff + half * 8;
    f.d4[0] = *(const uint4*)(c);
    f.d4[1] = *(const uint4*)(c + 16);
}

static __device__ __forceinline__ v8f wmma_bf16(const FragA& a, const FragB& b, v8f c) {
    return __builtin_amdgcn_wmma_f32_16x16x32_bf16(false, a.v, false, b.v,
                                                   (short)0, c, false, false);
}

// ---------------- conversion kernels ----------------

__global__ void cvt_x_wo(const float* __restrict__ x, const float* __restrict__ wo,
                         unsigned short* __restrict__ xb, unsigned short* __restrict__ wob) {
    size_t i = (size_t)blockIdx.x * blockDim.x + threadIdx.x;
    const size_t NX = (size_t)BT * CDIM;
    const size_t NW = (size_t)CDIM * CDIM;
    if (i < NX)              xb[i]       = f2bf(x[i]);
    else if (i < NX + NW)    wob[i - NX] = f2bf(wo[i - NX]);
}

// Wq/Wk/Wv [H][C][HS] fp32 -> transposed bf16 [H][HS][C] (column-major B operand)
__global__ void cvt_w_t(const float* __restrict__ wq, const float* __restrict__ wk,
                        const float* __restrict__ wv,
                        unsigned short* __restrict__ wqT, unsigned short* __restrict__ wkT,
                        unsigned short* __restrict__ wvT) {
    size_t i = (size_t)blockIdx.x * blockDim.x + threadIdx.x;
    const size_t NM = (size_t)NH * CDIM * HS;
    if (i >= 3 * NM) return;
    int    mat = (int)(i / NM);
    size_t j   = i % NM;
    int h   = (int)(j / ((size_t)CDIM * HS));
    int rem = (int)(j % ((size_t)CDIM * HS));
    int c = rem / HS, n = rem % HS;
    const float* src = (mat == 0) ? wq : (mat == 1) ? wk : wv;
    unsigned short* dst = (mat == 0) ? wqT : (mat == 1) ? wkT : wvT;
    dst[((size_t)h * HS + n) * CDIM + c] = f2bf(src[j]);
}

// ---------------- QKV projection ----------------
// One wave computes a 16x64 tile of q/k/v for one head. q,k stored [B,H,T,HS];
// v stored transposed [B,H,HS,T] so it can be a column-major B operand in P*V.

__global__ void qkv_gemm(const unsigned short* __restrict__ xb,
                         const unsigned short* __restrict__ wqT,
                         const unsigned short* __restrict__ wkT,
                         const unsigned short* __restrict__ wvT,
                         unsigned short* __restrict__ qb,
                         unsigned short* __restrict__ kb,
                         unsigned short* __restrict__ vTb) {
    const int gwid = (int)((blockIdx.x * blockDim.x + threadIdx.x) >> 5);
    const int rt   = gwid % (BT / 16);
    const int mh   = gwid / (BT / 16);
    const int h    = mh % NH;
    const int mat  = mh / NH;                 // 0=q 1=k 2=v (uniform per wave)
    const int lane = threadIdx.x & 31, half = lane >> 4, l16 = lane & 15;

    const unsigned short* wT    = (mat == 0) ? wqT : (mat == 1) ? wkT : wvT;
    const unsigned short* bcol  = wT + (size_t)h * HS * CDIM;       // [HS][C]
    const unsigned short* atile = xb + (size_t)rt * 16 * CDIM;

    v8f acc[4] = {v8f{0}, v8f{0}, v8f{0}, v8f{0}};
    for (int ks = 0; ks < CDIM; ks += 32) {
        FragA a; load_a(a, atile, CDIM, ks, l16, half);
#pragma unroll
        for (int nt = 0; nt < 4; ++nt) {
            FragB b; load_b(b, bcol + (size_t)(nt * 16) * CDIM, CDIM, ks, l16, half);
            acc[nt] = wmma_bf16(a, b, acc[nt]);
        }
    }

    const int row0 = rt * 16;
    const int bidx = row0 / TSEQ;             // uniform: 16 | TSEQ
#pragma unroll
    for (int nt = 0; nt < 4; ++nt) {
#pragma unroll
        for (int r = 0; r < 8; ++r) {
            const int rowm = r + 8 * half;
            const int t    = (row0 % TSEQ) + rowm;
            const int n    = nt * 16 + l16;
            const unsigned short val = f2bf(acc[nt][r]);
            if (mat == 2)
                vTb[(((size_t)bidx * NH + h) * HS + n) * TSEQ + t] = val;
            else if (mat == 0)
                qb[(((size_t)bidx * NH + h) * TSEQ + t) * HS + n] = val;
            else
                kb[(((size_t)bidx * NH + h) * TSEQ + t) * HS + n] = val;
        }
    }
}

// ---------------- flash attention core ----------------
// One wave per (b, h, 16-row query tile). Online softmax over 32-key tiles.

__global__ void flash_attn(const unsigned short* __restrict__ qbuf,
                           const unsigned short* __restrict__ kbuf,
                           const unsigned short* __restrict__ vTbuf,
                           unsigned short* __restrict__ attn) {
    __shared__ __align__(16) unsigned short pbuf[8][16 * 32];  // per-wave P tile
    const int wib  = threadIdx.x >> 5;
    const int gwid = (int)((blockIdx.x * blockDim.x + threadIdx.x) >> 5);
    const int qt   = gwid % (TSEQ / 16);
    const int bh   = gwid / (TSEQ / 16);
    const int lane = threadIdx.x & 31, half = lane >> 4, l16 = lane & 15;
    const int q0   = qt * 16;

    const unsigned short* qtile = qbuf  + ((size_t)bh * TSEQ + q0) * HS;
    const unsigned short* kmat  = kbuf  + (size_t)bh * TSEQ * HS;
    const unsigned short* vT    = vTbuf + (size_t)bh * HS * TSEQ;

    FragA qa0, qa1;
    load_a(qa0, qtile, HS, 0,  l16, half);
    load_a(qa1, qtile, HS, 32, l16, half);

    v8f o[4] = {v8f{0}, v8f{0}, v8f{0}, v8f{0}};
    float m[8], lsum[8];
#pragma unroll
    for (int r = 0; r < 8; ++r) { m[r] = -1e30f; lsum[r] = 0.0f; }

    const float scale = 0.03125f;            // C^-0.5 = 1/32
    const int kend = q0 + 16;                // causal: keys < kend
    for (int k0 = 0; k0 < kend; k0 += 32) {
        FragB kb00, kb01, kb10, kb11;        // K rows are column-major B for Q*K^T
        load_b(kb00, kmat + (size_t)k0        * HS, HS, 0,  l16, half);
        load_b(kb01, kmat + (size_t)k0        * HS, HS, 32, l16, half);
        load_b(kb10, kmat + (size_t)(k0 + 16) * HS, HS, 0,  l16, half);
        load_b(kb11, kmat + (size_t)(k0 + 16) * HS, HS, 32, l16, half);

        v8f s0 = v8f{0}, s1 = v8f{0};
        s0 = wmma_bf16(qa0, kb00, s0);  s0 = wmma_bf16(qa1, kb01, s0);
        s1 = wmma_bf16(qa0, kb10, s1);  s1 = wmma_bf16(qa1, kb11, s1);

        float a0[8], a1[8], mloc[8];
#pragma unroll
        for (int r = 0; r < 8; ++r) {
            const int row = q0 + r + 8 * half;
            const int c0  = k0 + l16;
            a0[r] = (c0      <= row) ? s0[r] * scale : -1e30f;
            a1[r] = (c0 + 16 <= row) ? s1[r] * scale : -1e30f;
            mloc[r] = fmaxf(a0[r], a1[r]);
        }
#pragma unroll
        for (int off = 1; off < 16; off <<= 1)
#pragma unroll
            for (int r = 0; r < 8; ++r)
                mloc[r] = fmaxf(mloc[r], __shfl_xor(mloc[r], off, 32));

        float alpha[8], rs[8];
#pragma unroll
        for (int r = 0; r < 8; ++r) {
            const float mn = fmaxf(m[r], mloc[r]);
            alpha[r] = __expf(m[r] - mn);
            m[r]     = mn;
            a0[r]    = __expf(a0[r] - mn);
            a1[r]    = __expf(a1[r] - mn);
            rs[r]    = a0[r] + a1[r];
        }
#pragma unroll
        for (int off = 1; off < 16; off <<= 1)
#pragma unroll
            for (int r = 0; r < 8; ++r)
                rs[r] += __shfl_xor(rs[r], off, 32);
#pragma unroll
        for (int r = 0; r < 8; ++r) lsum[r] = lsum[r] * alpha[r] + rs[r];
#pragma unroll
        for (int nt = 0; nt < 4; ++nt)
#pragma unroll
            for (int r = 0; r < 8; ++r) o[nt][r] *= alpha[r];

        // re-swizzle P (C/D layout) -> A fragment through wave-private LDS
        unsigned short* pw = pbuf[wib];
#pragma unroll
        for (int r = 0; r < 8; ++r) {
            const int rowm = r + 8 * half;
            pw[rowm * 32 + l16]      = f2bf(a0[r]);
            pw[rowm * 32 + 16 + l16] = f2bf(a1[r]);
        }
        asm volatile("s_wait_dscnt 0" ::: "memory");  // wave-local LDS RAW
        FragA pa;
        {
            const unsigned short* rdp = pw + l16 * 32 + half * 8;
            pa.d4[0] = *(const uint4*)(rdp);
            pa.d4[1] = *(const uint4*)(rdp + 16);
        }

#pragma unroll
        for (int nt = 0; nt < 4; ++nt) {
            FragB vb;  // V^T [HS][T] is the column-major B for P*V
            load_b(vb, vT + (size_t)(nt * 16) * TSEQ, TSEQ, k0, l16, half);
            o[nt] = wmma_bf16(pa, vb, o[nt]);
        }
    }

    const int b = bh / NH, h = bh % NH;
#pragma unroll
    for (int nt = 0; nt < 4; ++nt)
#pragma unroll
        for (int r = 0; r < 8; ++r) {
            const int row = q0 + r + 8 * half;
            attn[((size_t)(b * TSEQ + row)) * CDIM + h * HS + nt * 16 + l16] =
                f2bf(o[nt][r] / lsum[r]);
        }
}

// ---------------- output projection: out = attn @ Wo^T ----------------
// Row-major Wo is exactly the column-major B operand: B[k][n] = Wo[n*C + k].

__global__ void out_proj(const unsigned short* __restrict__ attn,
                         const unsigned short* __restrict__ wob,
                         float* __restrict__ out) {
    const int gwid  = (int)((blockIdx.x * blockDim.x + threadIdx.x) >> 5);
    const int ntile = gwid % (CDIM / 64);
    const int rt    = gwid / (CDIM / 64);
    const int lane  = threadIdx.x & 31, half = lane >> 4, l16 = lane & 15;

    const unsigned short* atile = attn + (size_t)rt * 16 * CDIM;
    v8f acc[4] = {v8f{0}, v8f{0}, v8f{0}, v8f{0}};
    for (int ks = 0; ks < CDIM; ks += 32) {
        FragA a; load_a(a, atile, CDIM, ks, l16, half);
#pragma unroll
        for (int nt = 0; nt < 4; ++nt) {
            FragB b; load_b(b, wob + (size_t)(ntile * 64 + nt * 16) * CDIM, CDIM, ks, l16, half);
            acc[nt] = wmma_bf16(a, b, acc[nt]);
        }
    }
#pragma unroll
    for (int nt = 0; nt < 4; ++nt)
#pragma unroll
        for (int r = 0; r < 8; ++r)
            out[((size_t)(rt * 16 + r + 8 * half)) * CDIM + ntile * 64 + nt * 16 + l16] =
                acc[nt][r];
}

// ---------------- launcher ----------------

extern "C" void kernel_launch(void* const* d_in, const int* in_sizes, int n_in,
                              void* d_out, int out_size, void* d_ws, size_t ws_size,
                              hipStream_t stream) {
    (void)in_sizes; (void)n_in; (void)out_size; (void)ws_size;
    const float* x  = (const float*)d_in[0];
    const float* wq = (const float*)d_in[1];
    const float* wk = (const float*)d_in[2];
    const float* wv = (const float*)d_in[3];
    const float* wo = (const float*)d_in[4];
    float* out = (float*)d_out;

    unsigned short* ws = (unsigned short*)d_ws;
    size_t off = 0;
    unsigned short* xb   = ws + off; off += (size_t)BT * CDIM;          // 4M
    unsigned short* wob  = ws + off; off += (size_t)CDIM * CDIM;        // 1M
    unsigned short* wqT  = ws + off; off += (size_t)NH * HS * CDIM;     // 1M
    unsigned short* wkT  = ws + off; off += (size_t)NH * HS * CDIM;     // 1M
    unsigned short* wvT  = ws + off; off += (size_t)NH * HS * CDIM;     // 1M
    unsigned short* qb   = ws + off; off += (size_t)BATCH * NH * TSEQ * HS; // 4M
    unsigned short* kbuf = ws + off; off += (size_t)BATCH * NH * TSEQ * HS; // 4M
    unsigned short* vTb  = ws + off; off += (size_t)BATCH * NH * HS * TSEQ; // 4M
    unsigned short* attn = ws + off; off += (size_t)BT * CDIM;          // 4M
    // total 24M ushorts = 48 MiB of workspace

    {   // x + Wo -> bf16
        size_t n = (size_t)BT * CDIM + (size_t)CDIM * CDIM;
        dim3 g((unsigned)((n + 255) / 256));
        cvt_x_wo<<<g, 256, 0, stream>>>(x, wo, xb, wob);
    }
    {   // Wq/Wk/Wv -> bf16 transposed [H][HS][C]
        size_t n = (size_t)3 * NH * CDIM * HS;
        dim3 g((unsigned)((n + 255) / 256));
        cvt_w_t<<<g, 256, 0, stream>>>(wq, wk, wv, wqT, wkT, wvT);
    }
    {   // QKV projection: 3 * H * (BT/16) waves, 8 waves per block
        int waves = 3 * NH * (BT / 16);
        qkv_gemm<<<waves / 8, 256, 0, stream>>>(xb, wqT, wkT, wvT, qb, kbuf, vTb);
    }
    {   // flash attention: B * H * (T/16) waves
        int waves = BATCH * NH * (TSEQ / 16);
        flash_attn<<<waves / 8, 256, 0, stream>>>(qb, kbuf, vTb, attn);
    }
    {   // output projection: (BT/16) * (C/64) waves
        int waves = (BT / 16) * (CDIM / 64);
        out_proj<<<waves / 8, 256, 0, stream>>>(attn, wob, out);
    }
}
